// FlowLMFullWrapper_74062416052620
// MI455X (gfx1250) — compile-verified, matrix operand order checked by reading, not run on hardware
//
#include <hip/hip_runtime.h>

#define B_SZ   32
#define T_TEXT 256
#define D_DIM  1024
#define F_FR   64
#define LDIM   32
#define LSD    8
#define DFF    4096
#define FH     1024

typedef __attribute__((ext_vector_type(16))) __bf16 bf16x16;
typedef __attribute__((ext_vector_type(8)))  __bf16 bf16x8;
typedef __attribute__((ext_vector_type(8)))  float  f32x8;

enum GemmMode { M_STORE = 0, M_BIAS = 1, M_BIAS_RELU = 2, M_RES = 3,
                M_BIAS_RES = 4, M_EULER_FH = 5, M_EULER_X = 6 };

// ---------------------------------------------------------------------------
// Weight pre-pass: W fp32 [Kd][Nc] row-major  ->  Wt bf16 [Nout][Kd] row-major
// (transposed, rows Nc..Nout-1 zero-filled). Runs once per call; weights are
// then reused 64..512x from L2 by the GEMMs with no inline conversion.
// ---------------------------------------------------------------------------
__global__ __launch_bounds__(256)
void wconv_kernel(const float* __restrict__ W, __bf16* __restrict__ Wt,
                  int Kd, int Nc, int Nout)
{
  __shared__ float tile[32][33];
  const int kb = blockIdx.x * 32;
  const int nb = blockIdx.y * 32;
  const int tid = threadIdx.x;
  const int r = tid >> 3, c4 = (tid & 7) << 2;
  #pragma unroll
  for (int u = 0; u < 4; ++u) {
    const int n = nb + c4 + u;
    tile[r][c4 + u] = (n < Nc) ? W[(size_t)(kb + r) * Nc + n] : 0.f;
  }
  __syncthreads();
  if (nb + r < Nout) {
    #pragma unroll
    for (int u = 0; u < 4; ++u)
      Wt[(size_t)(nb + r) * Kd + kb + c4 + u] = (__bf16)tile[c4 + u][r];
  }
}

// ---------------------------------------------------------------------------
// Tiled GEMM: C(M,N) = epilogue(A(M,K) @ B(K,N)).
// A: fp32 [M][K] (activations), converted to bf16 while staging (VALU path).
// B: bf16 [N][K] (pre-transposed weights, N padded to a multiple of 128),
//    staged with GLOBAL_LOAD_ASYNC_TO_LDS_B128 (raw bytes, ASYNCcnt-tracked).
// Block = 256 threads = 8 waves; macro-tile 32(M) x 128(N); K-step 32.
// Double-buffered LDS: slab k+1 is staged (async copy + A convert) while the
// WMMAs consume slab k; one s_wait_asynccnt + barrier per K-step.
// Each wave owns two 16x16 C tiles: one A fragment feeds two WMMAs.
// Fragments are contiguous ds_load_b128 pairs (K={0..7,16..23}+8*hi).
// Requires M%32==0, K%32==0; N guarded only at the store.
// ---------------------------------------------------------------------------
__global__ __launch_bounds__(256)
void wmma_gemm_kernel(const float* __restrict__ A, int lda,
                      const __bf16* __restrict__ Bt,
                      float* __restrict__ C, int ldc,
                      const float* __restrict__ Rin, int ldr,
                      const float* __restrict__ bias,
                      int N, int K, int mode, float scale,
                      const float* __restrict__ e0,
                      const float* __restrict__ e1,
                      const float* __restrict__ e2,
                      const float* __restrict__ sts,
                      const float* __restrict__ stt, int jidx)
{
  __shared__ __bf16 As[2][32][40];    // [buf][m][k], 80B pitch (16B aligned)
  __shared__ __bf16 Bs[2][128][40];   // [buf][n][k], 80B pitch

  const int tid  = threadIdx.x;
  const int m0   = blockIdx.y * 32;
  const int n0   = blockIdx.x * 128;
  const int wave = tid >> 5, lane = tid & 31;
  const int mt   = wave >> 2;             // 0..1
  const int ntb  = (wave & 3) * 2;        // 0,2,4,6 (two tiles per wave)
  const int hsel = lane >> 4, lrow = lane & 15;

  const int ar = tid >> 3, ac4 = (tid & 7) << 2;   // A stage: 32 rows x 8 f4

  const float* aptr = A + (size_t)(m0 + ar) * lda + ac4;

  f32x8 acc0 = {0.f, 0.f, 0.f, 0.f, 0.f, 0.f, 0.f, 0.f};
  f32x8 acc1 = {0.f, 0.f, 0.f, 0.f, 0.f, 0.f, 0.f, 0.f};

  // stage slab k0 into buffer b: A via VALU convert, B via async global->LDS
  auto stage = [&](int b, int k0) {
    float4 av = *(const float4*)(aptr + k0);
    As[b][ar][ac4 + 0] = (__bf16)av.x;  As[b][ar][ac4 + 1] = (__bf16)av.y;
    As[b][ar][ac4 + 2] = (__bf16)av.z;  As[b][ar][ac4 + 3] = (__bf16)av.w;
    #pragma unroll
    for (int i = 0; i < 2; ++i) {
      const int c  = tid + (i << 8);
      const int rB = c >> 2;               // 0..127
      const int c8 = (c & 3) << 3;         // 0,8,16,24 (bf16 elems)
      const __bf16* g = Bt + (size_t)(n0 + rB) * K + k0 + c8;
      const unsigned l = (unsigned)(size_t)&Bs[b][rB][c8];
      asm volatile("global_load_async_to_lds_b128 %0, %1, off"
                   :: "v"(l), "v"((unsigned long long)(size_t)g)
                   : "memory");
    }
  };

  // consume buffer b: build fragments, run two WMMAs
  auto compute = [&](int b) {
    const int arow = mt * 16 + lrow;
    bf16x8 alo = *(const bf16x8*)&As[b][arow][hsel * 8];
    bf16x8 ahi = *(const bf16x8*)&As[b][arow][16 + hsel * 8];
    bf16x16 afrag = __builtin_shufflevector(alo, ahi,
        0, 1, 2, 3, 4, 5, 6, 7, 8, 9, 10, 11, 12, 13, 14, 15);

    const int brow0 = ntb * 16 + lrow;
    bf16x8 b0lo = *(const bf16x8*)&Bs[b][brow0][hsel * 8];
    bf16x8 b0hi = *(const bf16x8*)&Bs[b][brow0][16 + hsel * 8];
    bf16x8 b1lo = *(const bf16x8*)&Bs[b][brow0 + 16][hsel * 8];
    bf16x8 b1hi = *(const bf16x8*)&Bs[b][brow0 + 16][16 + hsel * 8];
    bf16x16 bfrag0 = __builtin_shufflevector(b0lo, b0hi,
        0, 1, 2, 3, 4, 5, 6, 7, 8, 9, 10, 11, 12, 13, 14, 15);
    bf16x16 bfrag1 = __builtin_shufflevector(b1lo, b1hi,
        0, 1, 2, 3, 4, 5, 6, 7, 8, 9, 10, 11, 12, 13, 14, 15);

    acc0 = __builtin_amdgcn_wmma_f32_16x16x32_bf16(
        false, afrag, false, bfrag0, (short)0, acc0, false, false);
    acc1 = __builtin_amdgcn_wmma_f32_16x16x32_bf16(
        false, afrag, false, bfrag1, (short)0, acc1, false, false);
  };

  // software pipeline: prologue stage, then overlap stage(k+1) with compute(k)
  stage(0, 0);
  asm volatile("s_wait_asynccnt 0" ::: "memory");
  __syncthreads();

  int cur = 0;
  for (int k0 = 32; k0 < K; k0 += 32) {
    stage(cur ^ 1, k0);                       // fill other buffer
    if (k0 + 32 < K) __builtin_prefetch(aptr + k0 + 32, 0, 1);
    compute(cur);                             // consume current buffer
    asm volatile("s_wait_asynccnt 0" ::: "memory");
    __syncthreads();                          // next buffer ready; readers done
    cur ^= 1;
  }
  compute(cur);                               // epilogue slab

  // ---- epilogue ----------------------------------------------------------
  float sA = 0.f, sB = 0.f;
  if (mode == M_EULER_FH) { sA = sts[jidx]; sB = stt[jidx]; }
  #pragma unroll
  for (int u = 0; u < 2; ++u) {
    const int col = n0 + (ntb + u) * 16 + lrow;
    if (col >= N) continue;
    f32x8 a = u ? acc1 : acc0;
    #pragma unroll
    for (int v = 0; v < 8; ++v) {
      const int row = m0 + mt * 16 + hsel * 8 + v;
      float r = a[v];
      switch (mode) {
        case M_STORE:      break;
        case M_BIAS:       r += bias[col]; break;
        case M_BIAS_RELU:  r += bias[col]; r = fmaxf(r, 0.f); break;
        case M_RES:        r += Rin[(size_t)row * ldr + col]; break;
        case M_BIAS_RES:   r += bias[col] + Rin[(size_t)row * ldr + col]; break;
        case M_EULER_FH:   // relu(c@Wfc + x@Wfx + s*wfs + t*wft + bf1)
          r += Rin[(size_t)row * ldr + col] + sA * e0[col] + sB * e1[col] + e2[col];
          r = fmaxf(r, 0.f);
          break;
        case M_EULER_X:    // x + (fh@Wf2 + bf2) * dt
          r = Rin[(size_t)row * ldr + col] + (r + bias[col]) * scale;
          break;
      }
      C[(size_t)row * ldc + col] = r;
    }
  }
}

// ---------------------------------------------------------------------------
// Causal self-attention over the KV cache: one block per batch row.
// ---------------------------------------------------------------------------
__global__ __launch_bounds__(256)
void self_attn_kernel(const float* __restrict__ q, const float* __restrict__ ck,
                      const float* __restrict__ cv, float* __restrict__ ctx, int t)
{
  __shared__ float sc[F_FR];
  __shared__ float pr[F_FR];
  const int b = blockIdx.x, tid = threadIdx.x;
  const int wave = tid >> 5, lane = tid & 31;
  const float4* q4 = (const float4*)(q + (size_t)b * D_DIM);

  for (int f = wave; f <= t; f += 8) {
    const float4* k4 = (const float4*)(ck + ((size_t)b * F_FR + f) * D_DIM);
    float s = 0.f;
    for (int d = lane; d < D_DIM / 4; d += 32) {
      float4 a = q4[d], k = k4[d];
      s += a.x * k.x + a.y * k.y + a.z * k.z + a.w * k.w;
    }
    for (int o = 16; o > 0; o >>= 1) s += __shfl_xor(s, o, 32);
    if (lane == 0) sc[f] = s * 0.03125f;   // 1/sqrt(1024)
  }
  __syncthreads();
  if (tid == 0) {
    float mx = -1e30f;
    for (int f = 0; f <= t; ++f) mx = fmaxf(mx, sc[f]);
    float sum = 0.f;
    for (int f = 0; f <= t; ++f) { float e = __expf(sc[f] - mx); pr[f] = e; sum += e; }
    float inv = 1.f / sum;
    for (int f = 0; f <= t; ++f) pr[f] *= inv;
  }
  __syncthreads();
  for (int d = tid; d < D_DIM; d += 256) {
    float s = 0.f;
    for (int f = 0; f <= t; ++f) s += pr[f] * cv[((size_t)b * F_FR + f) * D_DIM + d];
    ctx[(size_t)b * D_DIM + d] = s;
  }
}

// ---------------------------------------------------------------------------
// Cross-attention to precomputed K_text/V_text: one block per batch row.
// ---------------------------------------------------------------------------
__global__ __launch_bounds__(256)
void cross_attn_kernel(const float* __restrict__ q2, const float* __restrict__ Kt,
                       const float* __restrict__ Vt, float* __restrict__ ctx)
{
  __shared__ float sc[T_TEXT];
  __shared__ float pr[T_TEXT];
  const int b = blockIdx.x, tid = threadIdx.x;
  const int wave = tid >> 5, lane = tid & 31;
  const float4* q4 = (const float4*)(q2 + (size_t)b * D_DIM);

  for (int tp = wave; tp < T_TEXT; tp += 8) {
    const float4* k4 = (const float4*)(Kt + ((size_t)b * T_TEXT + tp) * D_DIM);
    float s = 0.f;
    for (int d = lane; d < D_DIM / 4; d += 32) {
      float4 a = q4[d], k = k4[d];
      s += a.x * k.x + a.y * k.y + a.z * k.z + a.w * k.w;
    }
    for (int o = 16; o > 0; o >>= 1) s += __shfl_xor(s, o, 32);
    if (lane == 0) sc[tp] = s * 0.03125f;
  }
  __syncthreads();
  if (tid == 0) {
    float mx = -1e30f;
    for (int f = 0; f < T_TEXT; ++f) mx = fmaxf(mx, sc[f]);
    float sum = 0.f;
    for (int f = 0; f < T_TEXT; ++f) { float e = __expf(sc[f] - mx); pr[f] = e; sum += e; }
    float inv = 1.f / sum;
    for (int f = 0; f < T_TEXT; ++f) pr[f] *= inv;
  }
  __syncthreads();
  for (int d = tid; d < D_DIM; d += 256) {
    float s = 0.f;
    for (int f = 0; f < T_TEXT; ++f) s += pr[f] * Vt[((size_t)b * T_TEXT + f) * D_DIM + d];
    ctx[(size_t)b * D_DIM + d] = s;
  }
}

// eos[b, t] = c[b,:] . W_eos + b_eos
__global__ __launch_bounds__(256)
void eos_kernel(const float* __restrict__ c, const float* __restrict__ Weos,
                const float* __restrict__ beos, float* __restrict__ out, int t)
{
  __shared__ float red[256];
  const int b = blockIdx.x, tid = threadIdx.x;
  float s = 0.f;
  for (int d = tid; d < D_DIM; d += 256) s += c[(size_t)b * D_DIM + d] * Weos[d];
  red[tid] = s; __syncthreads();
  for (int o = 128; o > 0; o >>= 1) { if (tid < o) red[tid] += red[tid + o]; __syncthreads(); }
  if (tid == 0) out[(size_t)b * F_FR + t] = red[0] + beos[0];
}

__global__ void step_kernel(int* p)
{
  if (threadIdx.x == 0 && blockIdx.x == 0) *p = F_FR * (1 + T_TEXT);
}

// ---------------------------------------------------------------------------
extern "C" void kernel_launch(void* const* d_in, const int* in_sizes, int n_in,
                              void* d_out, int out_size, void* d_ws, size_t ws_size,
                              hipStream_t stream)
{
  (void)in_sizes; (void)n_in; (void)out_size; (void)ws_size;
  const float* text   = (const float*)d_in[0];
  const float* st_s   = (const float*)d_in[8];
  const float* st_t   = (const float*)d_in[9];
  const float* noise  = (const float*)d_in[10];
  const float* bos    = (const float*)d_in[11];
  const float* W_in   = (const float*)d_in[12];
  const float* b_in   = (const float*)d_in[13];
  const float* Wq     = (const float*)d_in[14];
  const float* Wk     = (const float*)d_in[15];
  const float* Wv     = (const float*)d_in[16];
  const float* Wo     = (const float*)d_in[17];
  const float* Wq2    = (const float*)d_in[18];
  const float* Wk2    = (const float*)d_in[19];
  const float* Wv2    = (const float*)d_in[20];
  const float* Wo2    = (const float*)d_in[21];
  const float* W_mlp1 = (const float*)d_in[22];
  const float* b_mlp1 = (const float*)d_in[23];
  const float* W_mlp2 = (const float*)d_in[24];
  const float* b_mlp2 = (const float*)d_in[25];
  const float* W_eos  = (const float*)d_in[26];
  const float* b_eos  = (const float*)d_in[27];
  const float* Wfc    = (const float*)d_in[28];
  const float* Wfx    = (const float*)d_in[29];
  const float* wfs    = (const float*)d_in[30];
  const float* wft    = (const float*)d_in[31];
  const float* bf1    = (const float*)d_in[32];
  const float* Wf2    = (const float*)d_in[33];
  const float* bf2    = (const float*)d_in[34];

  // Output regions (flat, in return order): lat, eos, ck_f, cv_f, step_f
  float* out = (float*)d_out;
  float* lat = out;
  float* eos = lat + (size_t)B_SZ * F_FR * LDIM;
  float* ckc = eos + (size_t)B_SZ * F_FR;
  float* cvc = ckc + (size_t)B_SZ * F_FR * D_DIM;
  int*   stp = (int*)(cvc + (size_t)B_SZ * F_FR * D_DIM);

  // Workspace: fp32 activations first, then bf16 transposed weights
  float* ws = (float*)d_ws;
  size_t o = 0;
  float* Ktext = ws + o; o += (size_t)B_SZ * T_TEXT * D_DIM;
  float* Vtext = ws + o; o += (size_t)B_SZ * T_TEXT * D_DIM;
  float* h0 = ws + o; o += (size_t)B_SZ * D_DIM;
  float* h1 = ws + o; o += (size_t)B_SZ * D_DIM;
  float* qb = ws + o; o += (size_t)B_SZ * D_DIM;
  float* cx = ws + o; o += (size_t)B_SZ * D_DIM;
  float* mb = ws + o; o += (size_t)B_SZ * DFF;
  float* cb = ws + o; o += (size_t)B_SZ * D_DIM;
  float* cf = ws + o; o += (size_t)B_SZ * FH;
  float* fh = ws + o; o += (size_t)B_SZ * FH;
  float* x0 = ws + o; o += (size_t)B_SZ * LDIM;
  float* x1 = ws + o; o += (size_t)B_SZ * LDIM;

  __bf16* wsb = (__bf16*)(ws + o);
  size_t ob = 0;
  __bf16* WinT  = wsb + ob; ob += (size_t)D_DIM * LDIM;     // [1024][32]
  __bf16* WqT   = wsb + ob; ob += (size_t)D_DIM * D_DIM;
  __bf16* WkT   = wsb + ob; ob += (size_t)D_DIM * D_DIM;
  __bf16* WvT   = wsb + ob; ob += (size_t)D_DIM * D_DIM;
  __bf16* WoT   = wsb + ob; ob += (size_t)D_DIM * D_DIM;
  __bf16* Wq2T  = wsb + ob; ob += (size_t)D_DIM * D_DIM;
  __bf16* Wk2T  = wsb + ob; ob += (size_t)D_DIM * D_DIM;
  __bf16* Wv2T  = wsb + ob; ob += (size_t)D_DIM * D_DIM;
  __bf16* Wo2T  = wsb + ob; ob += (size_t)D_DIM * D_DIM;
  __bf16* WfcT  = wsb + ob; ob += (size_t)FH * D_DIM;
  __bf16* Wm1T  = wsb + ob; ob += (size_t)DFF * D_DIM;      // [4096][1024]
  __bf16* Wm2T  = wsb + ob; ob += (size_t)D_DIM * DFF;      // [1024][4096]
  __bf16* WfxT  = wsb + ob; ob += (size_t)FH * LDIM;        // [1024][32]
  __bf16* Wf2T  = wsb + ob; ob += (size_t)128 * FH;         // [128][1024], rows 32..127 zero

  dim3 blk(256);

  #define WCONV(src, dst, Kd, Nc, Nout) \
    wconv_kernel<<<dim3((Kd) / 32, (Nout) / 32), blk, 0, stream>>>(src, dst, Kd, Nc, Nout)

  WCONV(W_in,   WinT, LDIM,  D_DIM, D_DIM);
  WCONV(Wq,     WqT,  D_DIM, D_DIM, D_DIM);
  WCONV(Wk,     WkT,  D_DIM, D_DIM, D_DIM);
  WCONV(Wv,     WvT,  D_DIM, D_DIM, D_DIM);
  WCONV(Wo,     WoT,  D_DIM, D_DIM, D_DIM);
  WCONV(Wq2,    Wq2T, D_DIM, D_DIM, D_DIM);
  WCONV(Wk2,    Wk2T, D_DIM, D_DIM, D_DIM);
  WCONV(Wv2,    Wv2T, D_DIM, D_DIM, D_DIM);
  WCONV(Wo2,    Wo2T, D_DIM, D_DIM, D_DIM);
  WCONV(Wfc,    WfcT, D_DIM, FH,    FH);
  WCONV(W_mlp1, Wm1T, D_DIM, DFF,   DFF);
  WCONV(W_mlp2, Wm2T, DFF,   D_DIM, D_DIM);
  WCONV(Wfx,    WfxT, LDIM,  FH,    FH);
  WCONV(Wf2,    Wf2T, FH,    LDIM,  128);
  #undef WCONV

  #define GEMM(Ap, lda, Bp, Cp, ldc, Rp, ldr, bp, Mm, Nn, Kk, md, sc, ee0, ee1, ee2, jj) \
    wmma_gemm_kernel<<<dim3(((Nn) + 127) / 128, (Mm) / 32), blk, 0, stream>>>( \
        Ap, lda, Bp, Cp, ldc, Rp, ldr, bp, Nn, Kk, md, sc, ee0, ee1, ee2, st_s, st_t, jj)
  #define NOF (const float*)nullptr

  // Static cross-attention K/V over text (the two big GEMMs)
  GEMM(text, D_DIM, Wk2T, Ktext, D_DIM, NOF, 0, NOF,
       B_SZ * T_TEXT, D_DIM, D_DIM, M_STORE, 0.f, NOF, NOF, NOF, 0);
  GEMM(text, D_DIM, Wv2T, Vtext, D_DIM, NOF, 0, NOF,
       B_SZ * T_TEXT, D_DIM, D_DIM, M_STORE, 0.f, NOF, NOF, NOF, 0);

  for (int t = 0; t < F_FR; ++t) {
    const float* seq = (t == 0) ? bos : (lat + (size_t)(t - 1) * LDIM);
    const int seqld  = (t == 0) ? 0 : F_FR * LDIM;   // lda=0 broadcasts bos_emb

    // h = seq @ W_in + b_in
    GEMM(seq, seqld, WinT, h0, D_DIM, NOF, 0, b_in,
         B_SZ, D_DIM, LDIM, M_BIAS, 0.f, NOF, NOF, NOF, 0);
    // q, k (-> cache), v (-> cache)
    GEMM(h0, D_DIM, WqT, qb, D_DIM, NOF, 0, NOF,
         B_SZ, D_DIM, D_DIM, M_STORE, 0.f, NOF, NOF, NOF, 0);
    GEMM(h0, D_DIM, WkT, ckc + (size_t)t * D_DIM, F_FR * D_DIM, NOF, 0, NOF,
         B_SZ, D_DIM, D_DIM, M_STORE, 0.f, NOF, NOF, NOF, 0);
    GEMM(h0, D_DIM, WvT, cvc + (size_t)t * D_DIM, F_FR * D_DIM, NOF, 0, NOF,
         B_SZ, D_DIM, D_DIM, M_STORE, 0.f, NOF, NOF, NOF, 0);
    self_attn_kernel<<<B_SZ, blk, 0, stream>>>(qb, ckc, cvc, cx, t);
    // h = h + ctx @ Wo
    GEMM(cx, D_DIM, WoT, h1, D_DIM, h0, D_DIM, NOF,
         B_SZ, D_DIM, D_DIM, M_RES, 0.f, NOF, NOF, NOF, 0);
    // cross attention
    GEMM(h1, D_DIM, Wq2T, qb, D_DIM, NOF, 0, NOF,
         B_SZ, D_DIM, D_DIM, M_STORE, 0.f, NOF, NOF, NOF, 0);
    cross_attn_kernel<<<B_SZ, blk, 0, stream>>>(qb, Ktext, Vtext, cx);
    GEMM(cx, D_DIM, Wo2T, h0, D_DIM, h1, D_DIM, NOF,
         B_SZ, D_DIM, D_DIM, M_RES, 0.f, NOF, NOF, NOF, 0);
    // FFN: c = h + relu(h@W_mlp1 + b1)@W_mlp2 + b2
    GEMM(h0, D_DIM, Wm1T, mb, DFF, NOF, 0, b_mlp1,
         B_SZ, DFF, D_DIM, M_BIAS_RELU, 0.f, NOF, NOF, NOF, 0);
    GEMM(mb, DFF, Wm2T, cb, D_DIM, h0, D_DIM, b_mlp2,
         B_SZ, D_DIM, DFF, M_BIAS_RES, 0.f, NOF, NOF, NOF, 0);
    eos_kernel<<<B_SZ, blk, 0, stream>>>(cb, W_eos, b_eos, eos, t);
    // hoisted Euler-loop invariant: cf = c @ Wfc
    GEMM(cb, D_DIM, WfcT, cf, FH, NOF, 0, NOF,
         B_SZ, FH, D_DIM, M_STORE, 0.f, NOF, NOF, NOF, 0);
    // LSD Euler integration
    for (int j = 0; j < LSD; ++j) {
      const float* xin; int xld;
      if (j == 0) { xin = noise + (size_t)t * LDIM; xld = F_FR * LDIM; }
      else        { xin = ((j - 1) & 1) ? x1 : x0;  xld = LDIM; }
      float* xout; int xoldc;
      if (j == LSD - 1) { xout = lat + (size_t)t * LDIM; xoldc = F_FR * LDIM; }
      else              { xout = (j & 1) ? x1 : x0;      xoldc = LDIM; }
      // fh = relu(cf + x@Wfx + st_s[j]*wfs + st_t[j]*wft + bf1)
      GEMM(xin, xld, WfxT, fh, FH, cf, FH, NOF,
           B_SZ, FH, LDIM, M_EULER_FH, 0.f, wfs, wft, bf1, j);
      // x = x + (fh@Wf2 + bf2) * dt   (Wf2T zero-padded to 128 rows)
      GEMM(fh, FH, Wf2T, xout, xoldc, xin, xld, bf2,
           B_SZ, LDIM, FH, M_EULER_X, 1.0f / (float)LSD, NOF, NOF, NOF, 0);
    }
  }
  step_kernel<<<1, 1, 0, stream>>>(stp);
  #undef GEMM
  #undef NOF
}
